// MLP_80221399155116
// MI455X (gfx1250) — matrix operand order repaired and checked by reading, not compiled, not run on hardware
//
#include <hip/hip_runtime.h>
#include <hip/hip_bf16.h>

typedef __attribute__((ext_vector_type(16))) __bf16 v16bf;
typedef __attribute__((ext_vector_type(8)))  __bf16 v8bf;
typedef __attribute__((ext_vector_type(8)))  float  v8f;

#define DIM   2048
#define VOCAB 32000
#define ROWS  4096      // B*S
#define SEQ   1024
#define NB    4
#define HID   100
#define LAB   5
#define INV_LN2 1.4426950408889634f

// ---------------- K1: hidden f32 -> Ab bf16 (elementwise) ----------------
__global__ __launch_bounds__(256) void cvtA_kernel(const float* __restrict__ h,
                                                   __bf16* __restrict__ Ab, int n) {
    int i = blockIdx.x * 256 + threadIdx.x;
    if (i < n) Ab[i] = (__bf16)h[i];
}

// ---------- K2: W[D][V] f32 -> Wt[V][D] bf16 (LDS tile transpose) ----------
__global__ __launch_bounds__(256) void cvtW_kernel(const float* __restrict__ W,
                                                   __bf16* __restrict__ Wt) {
    __shared__ float tile[32][33];
    int tv0 = blockIdx.x * 32;           // vocab tile
    int tk0 = blockIdx.y * 32;           // dim tile
    int tx = threadIdx.x & 31;
    int ty = threadIdx.x >> 5;           // 0..7
    #pragma unroll
    for (int i = 0; i < 32; i += 8)
        tile[ty + i][tx] = W[(size_t)(tk0 + ty + i) * VOCAB + tv0 + tx];
    __syncthreads();
    #pragma unroll
    for (int i = 0; i < 32; i += 8)
        Wt[(size_t)(tv0 + ty + i) * DIM + tk0 + tx] = (__bf16)tile[tx][ty + i];
}

// -------- K3: tok_logit[row] = dot(Ab[row], Wt[input_ids[row]]) --------
__global__ __launch_bounds__(256) void tok_kernel(const __bf16* __restrict__ Ab,
                                                  const __bf16* __restrict__ Wt,
                                                  const int* __restrict__ ids,
                                                  float* __restrict__ tok) {
    int row = blockIdx.x * 8 + (threadIdx.x >> 5);
    if (row >= ROWS) return;
    int lane = threadIdx.x & 31;
    int id = ids[row];
    const __bf16* a = Ab + (size_t)row * DIM;
    const __bf16* w = Wt + (size_t)id  * DIM;
    float acc = 0.0f;
    for (int k = lane; k < DIM; k += 32)
        acc += (float)a[k] * (float)w[k];
    #pragma unroll
    for (int off = 16; off > 0; off >>= 1)
        acc += __shfl_xor(acc, off);
    if (lane == 0) tok[row] = acc;
}

// ---- K4: per-row streaming logsumexp over VOCAB via bf16 WMMA tiles ----
// Grid: 128 blocks (32 rows = 2 row-tiles each) x 256 threads (8 waves).
// Wave w owns column tiles [w*250, (w+1)*250), processed 5 at a time.
// Each K-step: 2 A-fragment loads + 5 B-fragment loads feed 10 WMMAs
// (B reused across both row tiles, A reused across 5 column tiles).
// Softmax state is kept PER LANE (each lane owns fixed columns of C), so the
// hot loop has zero cross-lane ops; a single (m,s) butterfly merge runs at the end.
__global__ __launch_bounds__(256) void lse_kernel(const __bf16* __restrict__ Ab,
                                                  const __bf16* __restrict__ Wt,
                                                  const float* __restrict__ tok,
                                                  const int* __restrict__ mask,
                                                  float* __restrict__ surp) {
    __shared__ float lds_m[8][32];
    __shared__ float lds_s[8][32];

    const int row0 = blockIdx.x * 32;
    const int tid  = threadIdx.x;
    const int wave = tid >> 5;
    const int lane = tid & 31;
    const int half = lane >> 4;      // 0: lanes 0-15, 1: lanes 16-31
    const int mrow = lane & 15;      // A-fragment row / C-fragment column

    float m[2][8], s[2][8];
    #pragma unroll
    for (int t = 0; t < 2; ++t)
        #pragma unroll
        for (int r = 0; r < 8; ++r) { m[t][r] = -1e30f; s[t][r] = 0.0f; }

    // A fragments (16-bit A 16x32, contiguous K to match B: lane<16 -> K 0..15, lane>=16 -> K 16..31)
    const __bf16* aBase0 = Ab + (size_t)(row0 + mrow) * DIM + half * 16;
    const __bf16* aBase1 = aBase0 + (size_t)16 * DIM;
    const int tbase = wave * 250;

    for (int g = 0; g < 50; ++g) {
        v8f c[2][5];
        #pragma unroll
        for (int t = 0; t < 2; ++t)
            #pragma unroll
            for (int j = 0; j < 5; ++j)
                #pragma unroll
                for (int r = 0; r < 8; ++r) c[t][j][r] = 0.0f;

        const __bf16* bBase[5];
        #pragma unroll
        for (int j = 0; j < 5; ++j) {
            int v0 = (tbase + g * 5 + j) * 16;
            // B fragment (32x16 bf16): lane%16 = column, halves cover K 0..15 / 16..31
            bBase[j] = Wt + (size_t)(v0 + mrow) * DIM + half * 16;
        }

        for (int kk = 0; kk < DIM; kk += 32) {
            union { v16bf v; v8bf h[2]; } au0, au1;
            au0.h[0] = *(const v8bf*)(aBase0 + kk);
            au0.h[1] = *(const v8bf*)(aBase0 + kk + 8);
            au1.h[0] = *(const v8bf*)(aBase1 + kk);
            au1.h[1] = *(const v8bf*)(aBase1 + kk + 8);
            v16bf a0 = au0.v, a1 = au1.v;
            #pragma unroll
            for (int j = 0; j < 5; ++j) {
                v16bf b = *(const v16bf*)(bBase[j] + kk); // 16 contiguous K
                c[0][j] = __builtin_amdgcn_wmma_f32_16x16x32_bf16(
                              false, a0, false, b, (short)0, c[0][j], false, false);
                c[1][j] = __builtin_amdgcn_wmma_f32_16x16x32_bf16(
                              false, a1, false, b, (short)0, c[1][j], false, false);
            }
        }

        // Per-lane online logsumexp (lane owns column lane%16 of each tile;
        // VGPR r is row half*8+r of row-tile t). No cross-lane ops here.
        #pragma unroll
        for (int t = 0; t < 2; ++t)
            #pragma unroll
            for (int j = 0; j < 5; ++j)
                #pragma unroll
                for (int r = 0; r < 8; ++r) {
                    float x  = c[t][j][r];
                    float mn = fmaxf(m[t][r], x);
                    s[t][r]  = s[t][r] * __expf(m[t][r] - mn) + __expf(x - mn);
                    m[t][r]  = mn;
                }
    }

    // Merge (m,s) across the 16 lanes of each half-wave (columns of the tiles).
    #pragma unroll
    for (int t = 0; t < 2; ++t)
        #pragma unroll
        for (int r = 0; r < 8; ++r) {
            float mm = m[t][r], ss = s[t][r];
            #pragma unroll
            for (int msk = 1; msk <= 8; msk <<= 1) {
                float om = __shfl_xor(mm, msk);
                float os = __shfl_xor(ss, msk);
                float mn = fmaxf(mm, om);
                ss = ss * __expf(mm - mn) + os * __expf(om - mn);
                mm = mn;
            }
            m[t][r] = mm; s[t][r] = ss;
        }

    // half 0 tracks rows t*16 + 0..7, half 1 rows t*16 + 8..15 (replicated over 16 lanes)
    if ((lane & 15) == 0) {
        #pragma unroll
        for (int t = 0; t < 2; ++t)
            #pragma unroll
            for (int r = 0; r < 8; ++r) {
                lds_m[wave][t * 16 + half * 8 + r] = m[t][r];
                lds_s[wave][t * 16 + half * 8 + r] = s[t][r];
            }
    }
    __syncthreads();

    if (tid < 32) {
        float M = -1e30f;
        #pragma unroll
        for (int w = 0; w < 8; ++w) M = fmaxf(M, lds_m[w][tid]);
        float S = 0.0f;
        #pragma unroll
        for (int w = 0; w < 8; ++w) S += lds_s[w][tid] * __expf(lds_m[w][tid] - M);
        int row = row0 + tid;
        float lse = M + __logf(S);
        surp[row] = (lse - tok[row]) * INV_LN2 * (float)mask[row];
    }
}

// -------- K5: pooled[b][d] = mean_s feats; d==DIM slot is mean(surp) --------
__global__ __launch_bounds__(256) void pool_kernel(const float* __restrict__ hidden,
                                                   const float* __restrict__ surp,
                                                   float* __restrict__ pooled) {
    int i = blockIdx.x * 256 + threadIdx.x;
    if (i >= NB * (DIM + 1)) return;
    int b = i / (DIM + 1), d = i % (DIM + 1);
    float acc = 0.0f;
    if (d < DIM) {
        const float* p = hidden + (size_t)b * SEQ * DIM + d;
        for (int s2 = 0; s2 < SEQ; ++s2) acc += p[(size_t)s2 * DIM];
    } else {
        const float* p = surp + b * SEQ;
        for (int s2 = 0; s2 < SEQ; ++s2) acc += p[s2];
    }
    pooled[i] = acc * (1.0f / (float)SEQ);
}

// -------- K6: h = leaky_relu(pooled@Wr+br); out = [h,sent]@Wc+bc --------
__global__ __launch_bounds__(128) void head_kernel(const float* __restrict__ pooled,
                                                   const float* __restrict__ rw,
                                                   const float* __restrict__ rb,
                                                   const float* __restrict__ sent,
                                                   const float* __restrict__ cw,
                                                   const float* __restrict__ cb,
                                                   float* __restrict__ out) {
    __shared__ float h[NB][HID + 3];
    int t = threadIdx.x;
    for (int idx = t; idx < NB * HID; idx += 128) {
        int b = idx / HID, j = idx % HID;
        float acc = rb[j];
        const float* p = pooled + b * (DIM + 1);
        for (int k = 0; k < DIM + 1; ++k) acc += p[k] * rw[k * HID + j];
        h[b][j] = (acc > 0.0f) ? acc : 0.01f * acc;
    }
    if (t < NB * 3) h[t / 3][HID + (t % 3)] = sent[t];
    __syncthreads();
    if (t < NB * LAB) {
        int b = t / LAB, l = t % LAB;
        float acc = cb[l];
        for (int k = 0; k < HID + 3; ++k) acc += h[b][k] * cw[k * LAB + l];
        out[b * LAB + l] = acc;
    }
}

extern "C" void kernel_launch(void* const* d_in, const int* in_sizes, int n_in,
                              void* d_out, int out_size, void* d_ws, size_t ws_size,
                              hipStream_t stream) {
    const float* hidden = (const float*)d_in[0];
    const float* lm_w   = (const float*)d_in[1];
    const int*   ids    = (const int*)  d_in[2];
    const int*   amask  = (const int*)  d_in[3];
    const float* sent   = (const float*)d_in[4];
    const float* rw     = (const float*)d_in[5];
    const float* rb     = (const float*)d_in[6];
    const float* cw     = (const float*)d_in[7];
    const float* cb     = (const float*)d_in[8];
    float* out = (float*)d_out;

    char* ws = (char*)d_ws;
    __bf16* Ab = (__bf16*)ws;                                    // 16 MB
    __bf16* Wt = (__bf16*)(ws + (size_t)ROWS * DIM * 2);         // 131 MB
    float*  tok    = (float*)(ws + (size_t)ROWS * DIM * 2 + (size_t)VOCAB * DIM * 2);
    float*  surp   = tok + ROWS;
    float*  pooled = surp + ROWS;

    cvtA_kernel<<<(ROWS * DIM + 255) / 256, 256, 0, stream>>>(hidden, Ab, ROWS * DIM);
    cvtW_kernel<<<dim3(VOCAB / 32, DIM / 32), 256, 0, stream>>>(lm_w, Wt);
    tok_kernel<<<ROWS / 8, 256, 0, stream>>>(Ab, Wt, ids, tok);
    lse_kernel<<<ROWS / 32, 256, 0, stream>>>(Ab, Wt, tok, amask, surp);
    pool_kernel<<<(NB * (DIM + 1) + 255) / 256, 256, 0, stream>>>(hidden, surp, pooled);
    head_kernel<<<1, 128, 0, stream>>>(pooled, rw, rb, sent, cw, cb, out);
}
